// TransformerModel_4982162064023
// MI455X (gfx1250) — compile-verified, matrix operand order checked by reading, not run on hardware
//
#include <hip/hip_runtime.h>

// ---------------- model dims ----------------
#define VOCAB  10000
#define DMODEL 512
#define NHEAD  8
#define NLAYER 6
#define FFDIM  2048
#define SEQ    1024
#define BATCH  8
#define DKD    64
#define MROWS  (BATCH*SEQ)   // 8192 token rows

typedef __attribute__((ext_vector_type(16))) __bf16 v16bf;
typedef __attribute__((ext_vector_type(8)))  __bf16 v8bf;
typedef __attribute__((ext_vector_type(8)))  float  v8f;

// Load one WMMA 16-bit A/B fragment (16 bf16 per lane) for row base `row`,
// k-window k0..k0+31.  CDNA5 16-bit 16x32 layout: lane half h covers
// K = {8h..8h+7} in VGPR0..3 and K = {16+8h..23+8h} in VGPR4..7,
// i.e. two contiguous 16-byte runs.
__device__ __forceinline__ v16bf load_frag(const __bf16* row, int k0, int half) {
  union { v16bf v; v8bf h[2]; } u;
  u.h[0] = *(const v8bf*)(row + k0 + 8*half);
  u.h[1] = *(const v8bf*)(row + k0 + 16 + 8*half);
  return u.v;
}

#define WMMA_BF16(A_, B_, C_) \
  __builtin_amdgcn_wmma_f32_16x16x32_bf16(false, (A_), false, (B_), (short)0, (C_), false, false)

// ---------------- weight convert + transpose: wt[n*K+k] = (bf16)w[k*N+n] ---
__global__ void wconvT(const float* __restrict__ w, __bf16* __restrict__ wt,
                       int Kdim, int Ndim) {
  size_t idx = (size_t)blockIdx.x * blockDim.x + threadIdx.x;
  size_t total = (size_t)Kdim * Ndim;
  if (idx >= total) return;
  size_t n = idx / Kdim, k = idx % Kdim;
  wt[idx] = (__bf16)w[k * (size_t)Ndim + n];
}

// ---------------- embedding + sinusoidal positional encoding --------------
__global__ void embed_pos(const int* __restrict__ tok, const float* __restrict__ emb,
                          float* __restrict__ X, __bf16* __restrict__ Xb) {
  int row = blockIdx.x;            // 0..MROWS-1
  int s   = row & (SEQ - 1);
  int t   = tok[row];
  for (int d = threadIdx.x; d < DMODEL; d += blockDim.x) {
    // div = exp(-(2*(d/2)) * ln(10000)/D); ln(1e4)/512 = 0.017988946f
    float freq = __expf(-(float)(d & ~1) * 0.017988946039015984f);
    float ang  = (float)s * freq;
    float pe   = (d & 1) ? __cosf(ang) : __sinf(ang);
    float v = emb[(size_t)t * DMODEL + d] + pe;
    size_t idx = (size_t)row * DMODEL + d;
    X[idx]  = v;
    Xb[idx] = (__bf16)v;
  }
}

// ---------------- generic WMMA GEMM: C = A[M,K] x Bt[N,K]^T + bias --------
// 256 threads = 8 waves; each wave owns a 32x32 output tile as a 2x2 grid of
// 16x16 WMMA accumulators (each fragment feeds two WMMAs -> 2x reuse).
// Per-wave-uniform conditions are forced into SGPRs via readfirstlane so all
// WMMAs execute under scalar branches with EXEC all-1s (ISA requirement) and
// the k-loop load clause is never split by exec masking.
// outmode 0: row-major [M,N] (Cf fp32 and/or Cb bf16)
// outmode 1: Cb scattered to [B,H,S,DK]   (QK projections)
// outmode 2: Cb scattered to [B,H,DK,S]   (V projection, pre-transposed)
__global__ void gemm_wmma_bf16(const __bf16* __restrict__ A,
                               const __bf16* __restrict__ Bt,
                               const float*  __restrict__ bias,
                               float* Cf, __bf16* Cb,
                               int Mdim, int Ndim, int Kdim,
                               int relu, int outmode) {
  int wave = threadIdx.x >> 5;
  int lane = threadIdx.x & 31;
  int half = lane >> 4;
  int r    = lane & 15;
  int tm = blockIdx.y;                   // 32-row tile index
  int tn = blockIdx.x * 8 + wave;        // 32-col tile index
  int n0 = tn * 32;
  // wave-uniform conditions -> SGPR (scalar branch, EXEC untouched)
  if (__builtin_amdgcn_readfirstlane(n0 >= Ndim ? 1 : 0)) return;
  int hasN1 = __builtin_amdgcn_readfirstlane((n0 + 16) < Ndim ? 1 : 0);

  const __bf16* a0 = A  + (size_t)(tm * 32 +      r) * Kdim;
  const __bf16* a1 = A  + (size_t)(tm * 32 + 16 + r) * Kdim;
  const __bf16* b0 = Bt + (size_t)(n0      +      r) * Kdim;
  const __bf16* b1 = Bt + (size_t)(n0      + 16 + r) * Kdim;

  v8f acc00 = {}, acc01 = {}, acc10 = {}, acc11 = {};
  if (hasN1) {
    // full 2x2 tile: 8 b128 loads in one clause, 4 WMMAs per k-step
    for (int k0 = 0; k0 < Kdim; k0 += 32) {
      if (k0 + 64 < Kdim) {              // speculative prefetch of k+2 window
        __builtin_prefetch(a0 + k0 + 64, 0, 0);
        __builtin_prefetch(a1 + k0 + 64, 0, 0);
        __builtin_prefetch(b0 + k0 + 64, 0, 0);
        __builtin_prefetch(b1 + k0 + 64, 0, 0);
      }
      v16bf af0 = load_frag(a0, k0, half);
      v16bf af1 = load_frag(a1, k0, half);
      v16bf bf0 = load_frag(b0, k0, half);
      v16bf bf1 = load_frag(b1, k0, half);
      acc00 = WMMA_BF16(af0, bf0, acc00);
      acc10 = WMMA_BF16(af1, bf0, acc10);
      acc01 = WMMA_BF16(af0, bf1, acc01);
      acc11 = WMMA_BF16(af1, bf1, acc11);
    }
  } else {
    // boundary: last 16 columns only (N is always a multiple of 16)
    for (int k0 = 0; k0 < Kdim; k0 += 32) {
      if (k0 + 64 < Kdim) {
        __builtin_prefetch(a0 + k0 + 64, 0, 0);
        __builtin_prefetch(a1 + k0 + 64, 0, 0);
        __builtin_prefetch(b0 + k0 + 64, 0, 0);
      }
      v16bf af0 = load_frag(a0, k0, half);
      v16bf af1 = load_frag(a1, k0, half);
      v16bf bf0 = load_frag(b0, k0, half);
      acc00 = WMMA_BF16(af0, bf0, acc00);
      acc10 = WMMA_BF16(af1, bf0, acc10);
    }
  }

  auto store_tile = [&](v8f& acc, int mi, int ni) {
    int n = n0 + ni * 16 + r;
    float bvv = bias ? bias[n] : 0.0f;
#pragma unroll
    for (int j = 0; j < 8; ++j) {
      int m = tm * 32 + mi * 16 + j + 8 * half;  // D layout: VGPR j -> row j+8*half
      float v = acc[j] + bvv;
      if (relu) v = fmaxf(v, 0.0f);
      if (outmode == 0) {
        size_t idx = (size_t)m * Ndim + n;
        if (Cf) Cf[idx] = v;
        if (Cb) Cb[idx] = (__bf16)v;
      } else {
        int b = m >> 10, s = m & (SEQ - 1);      // m = b*SEQ + s (SEQ=1024)
        int h = n >> 6,  dk = n & (DKD - 1);
        size_t idx = (outmode == 1)
          ? (((size_t)(b * NHEAD + h)) * SEQ + s) * DKD + dk
          : (((size_t)(b * NHEAD + h)) * DKD + dk) * SEQ + s;
        Cb[idx] = (__bf16)v;
      }
    }
  };
  store_tile(acc00, 0, 0);
  store_tile(acc10, 1, 0);
  if (hasN1) {
    store_tile(acc01, 0, 1);
    store_tile(acc11, 1, 1);
  }
}

// ---------------- flash attention: one wave per (b,h, 16-query tile) ------
// Q,K: [B,H,S,DK] bf16; Vt: [B,H,DK,S] bf16; O: [B,S,D] bf16
__global__ void attn_flash(const __bf16* __restrict__ Q, const __bf16* __restrict__ Kc,
                           const __bf16* __restrict__ Vt, __bf16* __restrict__ O) {
  __shared__ float  sc[16][32];
  __shared__ __align__(32) __bf16 pb[16 * 32];
  __shared__ float  corr_s[16], rowm_s[16], rowl_s[16];

  int lane = threadIdx.x;                 // blockDim == 32 (one wave)
  int half = lane >> 4, r = lane & 15;
  int qt = blockIdx.x;                    // 0..SEQ/16-1
  int bh = blockIdx.y;                    // 0..B*H-1

  const __bf16* qrow = Q + ((size_t)bh * SEQ + qt * 16 + r) * DKD;
  v16bf qf0 = load_frag(qrow,  0, half);
  v16bf qf1 = load_frag(qrow, 32, half);

  if (lane < 16) { rowm_s[lane] = -1e30f; rowl_s[lane] = 0.0f; }
  __syncthreads();

  v8f o0 = {}, o1 = {}, o2 = {}, o3 = {};

  for (int kv = 0; kv < SEQ; kv += 32) {
    // --- scores S = Q(16x64) x K^T for 32 keys (two 16-col tiles) ---
#pragma unroll
    for (int t = 0; t < 2; ++t) {
      const __bf16* krow = Kc + ((size_t)bh * SEQ + kv + t * 16 + r) * DKD;
      v16bf kf0 = load_frag(krow,  0, half);
      v16bf kf1 = load_frag(krow, 32, half);
      v8f s = {};
      s = WMMA_BF16(qf0, kf0, s);
      s = WMMA_BF16(qf1, kf1, s);
#pragma unroll
      for (int j = 0; j < 8; ++j)
        sc[j + 8 * half][t * 16 + r] = s[j] * 0.125f;   // 1/sqrt(DK)
    }
    __syncthreads();

    // --- online softmax update, one lane per query row ---
    if (lane < 16) {
      float mold = rowm_s[lane];
      float mnew = mold;
      for (int c = 0; c < 32; ++c) mnew = fmaxf(mnew, sc[lane][c]);
      float cr  = __expf(mold - mnew);
      float sum = rowl_s[lane] * cr;
      for (int c = 0; c < 32; ++c) {
        float pv = __expf(sc[lane][c] - mnew);
        pb[lane * 32 + c] = (__bf16)pv;
        sum += pv;
      }
      rowm_s[lane] = mnew; rowl_s[lane] = sum; corr_s[lane] = cr;
    }
    __syncthreads();

    // --- rescale running O accumulators ---
#pragma unroll
    for (int j = 0; j < 8; ++j) {
      float cr = corr_s[j + 8 * half];
      o0[j] *= cr; o1[j] *= cr; o2[j] *= cr; o3[j] *= cr;
    }

    // --- O += P(16x32) x V(32x64): 4 column tiles, Vt rows contiguous ---
    v16bf pf = load_frag((const __bf16*)pb + r * 32, 0, half);
#pragma unroll
    for (int t = 0; t < 4; ++t) {
      const __bf16* vrow = Vt + ((size_t)bh * DKD + t * 16 + r) * SEQ + kv;
      v16bf vf = load_frag(vrow, 0, half);
      v8f* po = (t == 0) ? &o0 : (t == 1) ? &o1 : (t == 2) ? &o2 : &o3;
      *po = WMMA_BF16(pf, vf, *po);
    }
    __syncthreads();
  }

  // --- finalize: divide by softmax denom, scatter to [B,S,D] ---
  int b = bh >> 3, h = bh & 7;
#pragma unroll
  for (int j = 0; j < 8; ++j) {
    int m = qt * 16 + j + 8 * half;
    float inv = 1.0f / rowl_s[j + 8 * half];
    size_t base = ((size_t)b * SEQ + m) * DMODEL + (size_t)h * DKD;
    O[base +  0 + r] = (__bf16)(o0[j] * inv);
    O[base + 16 + r] = (__bf16)(o1[j] * inv);
    O[base + 32 + r] = (__bf16)(o2[j] * inv);
    O[base + 48 + r] = (__bf16)(o3[j] * inv);
  }
}

// ---------------- residual add + layernorm (one block per token row) ------
__global__ void residual_ln(const float* __restrict__ X, const float* __restrict__ Y,
                            const float* __restrict__ g, const float* __restrict__ bta,
                            float* __restrict__ Xo, __bf16* __restrict__ Xb) {
  __shared__ float red[256];
  int row = blockIdx.x, t = threadIdx.x;
  size_t base = (size_t)row * DMODEL;
  float a0 = X[base + t]       + Y[base + t];
  float a1 = X[base + t + 256] + Y[base + t + 256];
  red[t] = a0 + a1;
  __syncthreads();
  for (int s = 128; s > 0; s >>= 1) { if (t < s) red[t] += red[t + s]; __syncthreads(); }
  float mean = red[0] * (1.0f / DMODEL);
  __syncthreads();
  float d0 = a0 - mean, d1 = a1 - mean;
  red[t] = d0 * d0 + d1 * d1;
  __syncthreads();
  for (int s = 128; s > 0; s >>= 1) { if (t < s) red[t] += red[t + s]; __syncthreads(); }
  float rstd = rsqrtf(red[0] * (1.0f / DMODEL) + 1e-5f);
  float out0 = d0 * rstd * g[t]       + bta[t];
  float out1 = d1 * rstd * g[t + 256] + bta[t + 256];
  Xo[base + t]       = out0;          Xo[base + t + 256] = out1;
  Xb[base + t]       = (__bf16)out0;  Xb[base + t + 256] = (__bf16)out1;
}

// ---------------- host orchestration --------------------------------------
extern "C" void kernel_launch(void* const* d_in, const int* in_sizes, int n_in,
                              void* d_out, int out_size, void* d_ws, size_t ws_size,
                              hipStream_t stream) {
  const int*   tokens = (const int*)  d_in[0];
  const float* emb    = (const float*)d_in[1];
  const float* Wq     = (const float*)d_in[2];
  const float* bq     = (const float*)d_in[3];
  const float* Wk     = (const float*)d_in[4];
  const float* bk     = (const float*)d_in[5];
  const float* Wv     = (const float*)d_in[6];
  const float* bv     = (const float*)d_in[7];
  const float* Wo     = (const float*)d_in[8];
  const float* bo     = (const float*)d_in[9];
  const float* ln1s   = (const float*)d_in[10];
  const float* ln1b   = (const float*)d_in[11];
  const float* W1     = (const float*)d_in[12];
  const float* b1     = (const float*)d_in[13];
  const float* W2     = (const float*)d_in[14];
  const float* b2     = (const float*)d_in[15];
  const float* ln2s   = (const float*)d_in[16];
  const float* ln2b   = (const float*)d_in[17];
  const float* Wout   = (const float*)d_in[18];
  const float* bout   = (const float*)d_in[19];

  // ---- carve workspace (~157 MB total) ----
  char* p = (char*)d_ws;
  auto carve = [&](size_t elems, size_t elsz) -> void* {
    void* r = (void*)p;
    p += (elems * elsz + 255) & ~(size_t)255;
    return r;
  };
  const size_t DD = (size_t)DMODEL * DMODEL, DF = (size_t)DMODEL * FFDIM;
  __bf16* wqT   = (__bf16*)carve((size_t)NLAYER * DD, 2);
  __bf16* wkT   = (__bf16*)carve((size_t)NLAYER * DD, 2);
  __bf16* wvT   = (__bf16*)carve((size_t)NLAYER * DD, 2);
  __bf16* woT   = (__bf16*)carve((size_t)NLAYER * DD, 2);
  __bf16* w1T   = (__bf16*)carve((size_t)NLAYER * DF, 2);
  __bf16* w2T   = (__bf16*)carve((size_t)NLAYER * DF, 2);
  __bf16* woutT = (__bf16*)carve((size_t)DMODEL * VOCAB, 2);
  float*  Xf    = (float*) carve((size_t)MROWS * DMODEL, 4);
  __bf16* Xb    = (__bf16*)carve((size_t)MROWS * DMODEL, 2);
  float*  Yf    = (float*) carve((size_t)MROWS * DMODEL, 4);
  __bf16* Qb    = (__bf16*)carve((size_t)MROWS * DMODEL, 2);
  __bf16* Kb    = (__bf16*)carve((size_t)MROWS * DMODEL, 2);
  __bf16* Vtb   = (__bf16*)carve((size_t)MROWS * DMODEL, 2);
  __bf16* Ob    = (__bf16*)carve((size_t)MROWS * DMODEL, 2);
  __bf16* Hb    = (__bf16*)carve((size_t)MROWS * FFDIM,  2);

  // ---- weights -> bf16 transposed ----
  int gDD = (int)((DD + 255) / 256), gDF = (int)((DF + 255) / 256);
  for (int i = 0; i < NLAYER; ++i) {
    wconvT<<<gDD, 256, 0, stream>>>(Wq + i * DD, wqT + i * DD, DMODEL, DMODEL);
    wconvT<<<gDD, 256, 0, stream>>>(Wk + i * DD, wkT + i * DD, DMODEL, DMODEL);
    wconvT<<<gDD, 256, 0, stream>>>(Wv + i * DD, wvT + i * DD, DMODEL, DMODEL);
    wconvT<<<gDD, 256, 0, stream>>>(Wo + i * DD, woT + i * DD, DMODEL, DMODEL);
    wconvT<<<gDF, 256, 0, stream>>>(W1 + i * DF, w1T + i * DF, DMODEL, FFDIM);
    wconvT<<<gDF, 256, 0, stream>>>(W2 + i * DF, w2T + i * DF, FFDIM, DMODEL);
  }
  wconvT<<<(int)(((size_t)DMODEL * VOCAB + 255) / 256), 256, 0, stream>>>(
      Wout, woutT, DMODEL, VOCAB);

  embed_pos<<<MROWS, 256, 0, stream>>>(tokens, emb, Xf, Xb);

  auto gemm = [&](const __bf16* A, const __bf16* Bt, const float* bias,
                  float* Cf, __bf16* Cb, int M, int N, int K, int relu, int mode) {
    dim3 grid((N + 255) / 256, M / 32);   // 8 waves x (32x32) = 32 rows x 256 cols / block
    gemm_wmma_bf16<<<grid, 256, 0, stream>>>(A, Bt, bias, Cf, Cb, M, N, K, relu, mode);
  };

  for (int i = 0; i < NLAYER; ++i) {
    const float* bqi = bq + (size_t)i * DMODEL;
    const float* bki = bk + (size_t)i * DMODEL;
    const float* bvi = bv + (size_t)i * DMODEL;
    const float* boi = bo + (size_t)i * DMODEL;

    gemm(Xb, wqT + i * DD, bqi, nullptr, Qb,  MROWS, DMODEL, DMODEL, 0, 1);
    gemm(Xb, wkT + i * DD, bki, nullptr, Kb,  MROWS, DMODEL, DMODEL, 0, 1);
    gemm(Xb, wvT + i * DD, bvi, nullptr, Vtb, MROWS, DMODEL, DMODEL, 0, 2);

    attn_flash<<<dim3(SEQ / 16, BATCH * NHEAD), 32, 0, stream>>>(Qb, Kb, Vtb, Ob);

    gemm(Ob, woT + i * DD, boi, Yf, nullptr, MROWS, DMODEL, DMODEL, 0, 0);
    residual_ln<<<MROWS, 256, 0, stream>>>(Xf, Yf, ln1s + (size_t)i * DMODEL,
                                           ln1b + (size_t)i * DMODEL, Xf, Xb);

    gemm(Xb, w1T + i * DF, b1 + (size_t)i * FFDIM,  nullptr, Hb, MROWS, FFDIM,  DMODEL, 1, 0);
    gemm(Hb, w2T + i * DF, b2 + (size_t)i * DMODEL, Yf, nullptr, MROWS, DMODEL, FFDIM,  0, 0);
    residual_ln<<<MROWS, 256, 0, stream>>>(Xf, Yf, ln2s + (size_t)i * DMODEL,
                                           ln2b + (size_t)i * DMODEL, Xf, Xb);
  }

  gemm(Xb, woutT, bout, (float*)d_out, nullptr, MROWS, VOCAB, DMODEL, 0, 0);
}
